// Blender_70334384439403
// MI455X (gfx1250) — matrix operand order repaired and checked by reading, not compiled, not run on hardware
//
#include <hip/hip_runtime.h>
#include <math.h>
#include <stdint.h>

typedef __attribute__((ext_vector_type(16))) _Float16 v16h;
typedef __attribute__((ext_vector_type(8)))  _Float16 v8h;
typedef __attribute__((ext_vector_type(8)))  float    v8f;

#define NSAMP   64
#define CCH     128
#define HW      784            // 28*28
#define NTILES  49             // HW / 16
#define KSTEP   32             // K per WMMA
#define KCHUNKS (CCH / KSTEP)  // 4
#define WAVES_PER_BLOCK 8
#define HBLOCKS ((NTILES + WAVES_PER_BLOCK - 1) / WAVES_PER_BLOCK)  // 7

// ---------------------------------------------------------------------------
// Zero the two group accumulators (s_real, s_fake)
// ---------------------------------------------------------------------------
__global__ void bl_zero_accum(float* __restrict__ acc) {
    acc[0] = 0.0f;
    acc[1] = 0.0f;
}

// ---------------------------------------------------------------------------
// Per-pixel L2 normalize over channels, write fp16 features as [N, HW, C]
// row-major (each row = one pixel's channel vector).
// ---------------------------------------------------------------------------
__global__ __launch_bounds__(256)
void bl_normalize(const float* __restrict__ feats, _Float16* __restrict__ out) {
    const int n = blockIdx.x;
    const int p = blockIdx.y * 256 + threadIdx.x;
    if (p >= HW) return;

    const float* base = feats + (size_t)n * CCH * HW + p;
    float ss = 0.0f;
#pragma unroll 8
    for (int c = 0; c < CCH; ++c) {
        float v = base[(size_t)c * HW];
        ss += v * v;
    }
    const float inv = 1.0f / fmaxf(sqrtf(ss), 1e-12f);

    _Float16* o = out + ((size_t)n * HW + p) * CCH;
#pragma unroll 8
    for (int c = 0; c < CCH; ++c) {
        o[c] = (_Float16)(base[(size_t)c * HW] * inv);
    }
}

// Async global -> LDS copy of 16 bytes per lane (ASYNCcnt-tracked DMA path).
__device__ __forceinline__ void bl_async_copy16(uint32_t lds_off,
                                                const _Float16* gptr) {
    asm volatile("global_load_async_to_lds_b128 %0, %1, off"
                 :: "v"(lds_off), "v"((unsigned long long)(uintptr_t)gptr)
                 : "memory");
}

// ---------------------------------------------------------------------------
// One block = 8 waves = 8 h-strips of one sample's Gram matrix.
// B tiles (16x128 fp16, 4KB) are DMA'd global->LDS with
// global_load_async_to_lds_b128, double-buffered so tile t+1 streams in
// while tile t feeds 4x v_wmma_f32_16x16x32_f16 per wave. All four B
// fragments are preloaded from LDS in one ds clause (single s_wait_dscnt),
// then the four WMMAs issue back-to-back behind a sched_barrier.
//
// A layout (16-bit, 16x32): lane<16 -> M=lane, halves = K{k0..k0+7, k0+16..k0+23}
//                           lane>=16 -> M=lane-16, halves = K{k0+8.., k0+24..}
// B layout (16-bit, 32x16): lane<16 -> N=lane, halves = K{k0..k0+7, k0+16..k0+23}
//                           lane>=16 -> N=lane-16, halves = K{k0+8.., k0+24..}
// C/D layout: lane<16 -> (M=r, N=lane), lane>=16 -> (M=r+8, N=lane-16)
// ---------------------------------------------------------------------------
__global__ __launch_bounds__(256)
void bl_gram_expsum(const _Float16* __restrict__ F,      // [NSAMP, HW, CCH] fp16
                    const float* __restrict__ maskp,     // [NSAMP, HW]
                    const float* __restrict__ maskn,     // [NSAMP, HW]
                    float invT, float* __restrict__ accum) {
    __shared__ _Float16 lbuf[2][16 * CCH];               // 2 x 4KB B tiles

    const int n     = blockIdx.y;                        // sample
    const int wave  = threadIdx.x >> 5;
    const int lane  = threadIdx.x & 31;
    const int grp   = lane >> 4;                         // lane half: 0/1
    const int l16   = lane & 15;
    const int htile = blockIdx.x * WAVES_PER_BLOCK + wave;
    const bool active = (htile < NTILES);

    const _Float16* Fn = F + (size_t)n * HW * CCH;

    // Cooperative async copy of one B tile: 256 threads x 16B = 4KB.
    const int crow   = threadIdx.x >> 4;                 // 0..15
    const int cchunk = threadIdx.x & 15;                 // 0..15
    const int celem  = crow * CCH + cchunk * 8;          // element within tile

    // ---- A fragments for this wave's 16 rows, full K=128, kept resident ----
    v16h afrag[KCHUNKS];
    float mp[8];
    if (active) {
        const int h0 = htile * 16;
        const _Float16* rp = Fn + (size_t)(h0 + l16) * CCH + grp * 8;
#pragma unroll
        for (int k = 0; k < KCHUNKS; ++k) {
            v8h lo = *(const v8h*)(rp + k * KSTEP);
            v8h hi = *(const v8h*)(rp + k * KSTEP + 16);
            v16h a;
#pragma unroll
            for (int i = 0; i < 8; ++i) { a[i] = lo[i]; a[8 + i] = hi[i]; }
            afrag[k] = a;
        }
#pragma unroll
        for (int r = 0; r < 8; ++r)
            mp[r] = maskp[(size_t)n * HW + h0 + (r + 8 * grp)];
    }

    // Prologue: start DMA of tile 0 into buffer 0.
    bl_async_copy16((uint32_t)(uintptr_t)&lbuf[0][celem], Fn + celem);

    float acc = 0.0f;
    for (int t = 0; t < NTILES; ++t) {
        // Prefetch tile t+1 into the other buffer, then wait for tile t only
        // (ASYNCcnt is in-order per wave: <=1 outstanding means t is done).
        if (t + 1 < NTILES) {
            bl_async_copy16((uint32_t)(uintptr_t)&lbuf[(t + 1) & 1][celem],
                            Fn + (size_t)(t + 1) * 16 * CCH + celem);
            asm volatile("s_wait_asynccnt 0x1" ::: "memory");
        } else {
            asm volatile("s_wait_asynccnt 0x0" ::: "memory");
        }
        __syncthreads();                                 // tile t visible to all

        if (active) {
            const _Float16* lb = lbuf[t & 1];

            // Preload all four B fragments (8x ds_load_b128, one wait).
            v16h bfrag[KCHUNKS];
#pragma unroll
            for (int k = 0; k < KCHUNKS; ++k) {
                const _Float16* bp = lb + l16 * CCH + k * KSTEP + grp * 8;
                v8h lo = *(const v8h*)bp;
                v8h hi = *(const v8h*)(bp + 16);
                v16h b;
#pragma unroll
                for (int i = 0; i < 8; ++i) { b[i] = lo[i]; b[8 + i] = hi[i]; }
                bfrag[k] = b;
            }
            __builtin_amdgcn_sched_barrier(0);           // loads stay ahead of WMMAs

            v8f cacc = {};
#pragma unroll
            for (int k = 0; k < KCHUNKS; ++k) {
                cacc = __builtin_amdgcn_wmma_f32_16x16x32_f16(
                    false, afrag[k], false, bfrag[k], (short)0, cacc, false, false);
            }

            const float mn = maskn[(size_t)n * HW + t * 16 + l16];
#pragma unroll
            for (int r = 0; r < 8; ++r) {
                // masked-out pairs contribute exp(0) = 1
                acc += (mp[r] * mn > 0.5f) ? __expf(cacc[r] * invT) : 1.0f;
            }
        }
        __syncthreads();                                 // reads done before overwrite
    }

    if (active) {
#pragma unroll
        for (int off = 16; off > 0; off >>= 1)
            acc += __shfl_xor(acc, off, 32);
        if (lane == 0) atomicAdd(accum, acc);
    }
}

// ---------------------------------------------------------------------------
// out = -log(s_real / (s_fake + s_real))
// ---------------------------------------------------------------------------
__global__ void bl_finalize(const float* __restrict__ acc, float* __restrict__ out) {
    const float sr = acc[0];
    const float sf = acc[1];
    out[0] = -logf(sr / (sf + sr));
}

// ---------------------------------------------------------------------------
extern "C" void kernel_launch(void* const* d_in, const int* in_sizes, int n_in,
                              void* d_out, int out_size, void* d_ws, size_t ws_size,
                              hipStream_t stream) {
    (void)in_sizes; (void)n_in; (void)out_size; (void)ws_size;

    const float* real_feats = (const float*)d_in[0];
    const float* fake_feats = (const float*)d_in[1];
    const float* real_pos   = (const float*)d_in[2];
    const float* real_neg   = (const float*)d_in[3];
    const float* fake_pos   = (const float*)d_in[4];
    const float* fake_neg   = (const float*)d_in[5];
    float* out = (float*)d_out;

    // Workspace layout: [0,256) accumulators; then fp16 feature buffers.
    float*    acc   = (float*)d_ws;
    _Float16* Freal = (_Float16*)((char*)d_ws + 256);
    _Float16* Ffake = Freal + (size_t)NSAMP * HW * CCH;

    bl_zero_accum<<<1, 1, 0, stream>>>(acc);

    dim3 ngrid(NSAMP, (HW + 255) / 256);
    bl_normalize<<<ngrid, 256, 0, stream>>>(real_feats, Freal);
    bl_normalize<<<ngrid, 256, 0, stream>>>(fake_feats, Ffake);

    const float invT = 1.0f / 0.7f;
    dim3 ggrid(HBLOCKS, NSAMP);
    bl_gram_expsum<<<ggrid, 256, 0, stream>>>(Freal, real_pos, real_neg, invT, acc + 0);
    bl_gram_expsum<<<ggrid, 256, 0, stream>>>(Ffake, fake_pos, fake_neg, invT, acc + 1);

    bl_finalize<<<1, 1, 0, stream>>>(acc, out);
}
